// EncoderBlock_41300405518492
// MI455X (gfx1250) — compile-verified, matrix operand order checked by reading, not run on hardware
//
#include <hip/hip_runtime.h>
#include <hip/hip_bf16.h>

// ---------------------------------------------------------------------------
// Types for CDNA5 WMMA (wave32): bf16 A/B fragments (32B), f32 accumulators.
// ---------------------------------------------------------------------------
typedef __bf16 bf16_t;
typedef bf16_t v16bf __attribute__((ext_vector_type(16)));
typedef float  v8f   __attribute__((ext_vector_type(8)));
typedef unsigned int u32x4 __attribute__((ext_vector_type(4)));
typedef unsigned int u32x8 __attribute__((ext_vector_type(8)));

union FragBF {
  u32x4 q[2];   // two 16B chunks: k in [kh, kh+8) and [kh+16, kh+24)
  v16bf v;
};

__device__ __forceinline__ unsigned short f32_to_bf16_bits(float f) {
  unsigned int u = __float_as_uint(f);
  u += 0x7FFFu + ((u >> 16) & 1u);   // round-to-nearest-even
  return (unsigned short)(u >> 16);
}

__device__ __forceinline__ float gelu_exact(float x) {
  return 0.5f * x * (1.0f + erff(x * 0.70710678118654752f));
}

// LDS byte offset of a shared-memory object: per CDNA5 aperture decode,
// the LDS address is the low 32 bits of the generic (flat) pointer.
__device__ __forceinline__ unsigned lds_off_u32(const void* p) {
  return (unsigned)(unsigned long long)(uintptr_t)p;
}

// ---------------------------------------------------------------------------
// Tensor Data Mover: DMA one 2-D tile [rows=128 x 64B] from global -> LDS.
// Descriptor per CDNA5 ISA (D# groups 0/1, 2-D form,
// VADDR2/3 = NULL). data_size = 4B; tile_dim0 = 16 dwords (= 32 bf16 = 64B),
// tile_dim1 = 128 rows; tensor_dim0_stride = row_dw (K in dwords).
// workgroup_mask = 0 (not launched in a cluster). Tracked by TENSORcnt.
// ---------------------------------------------------------------------------
__device__ __forceinline__ void tdm_load_tile(unsigned lds_addr,
                                              const unsigned short* g,
                                              unsigned row_dw, unsigned rows) {
  unsigned long long ga = (unsigned long long)(uintptr_t)g;
  union { unsigned w[4]; u32x4 v; } g0;
  union { unsigned w[8]; u32x8 v; } g1;
  g0.w[0] = 1u;                                  // count=1, user descriptor
  g0.w[1] = lds_addr;                            // lds_addr[31:0]
  g0.w[2] = (unsigned)ga;                        // global_addr[31:0]
  g0.w[3] = (unsigned)((ga >> 32) & 0x01FFFFFFu) // global_addr[56:32]
          | (2u << 30);                          // type=2 ("image")
  g1.w[0] = (2u << 16);                          // wg_mask=0, data_size=4B
  g1.w[1] = (row_dw & 0xFFFFu) << 16;            // tensor_dim0[15:0]
  g1.w[2] = (row_dw >> 16) | ((rows & 0xFFFFu) << 16);  // dim0[31:16]|dim1[15:0]
  g1.w[3] = (rows >> 16) | (16u << 16);          // dim1[31:16] | tile_dim0=16
  g1.w[4] = 128u;                                // tile_dim1=128, tile_dim2=0
  g1.w[5] = row_dw;                              // tensor_dim0_stride[31:0]
  g1.w[6] = 0u;                                  // stride[47:32], dim1_stride lo
  g1.w[7] = 0u;
  asm volatile("tensor_load_to_lds %0, %1" :: "s"(g0.v), "s"(g1.v) : "memory");
}

// ---------------------------------------------------------------------------
// fp32 -> bf16 (bit pattern) elementwise
// ---------------------------------------------------------------------------
__global__ void cvt_kernel(const float* __restrict__ in,
                           unsigned short* __restrict__ out, int n) {
  int i = blockIdx.x * blockDim.x + threadIdx.x;
  if (i < n) out[i] = f32_to_bf16_bits(in[i]);
}

// ---------------------------------------------------------------------------
// Tiled bf16 WMMA GEMM: C[M,N] = act(A[M,K] * W[N,K]^T + bias[N])
// Block: 256 threads (8 waves). Tile: 128(M) x 128(N), K-step 32.
// Each wave computes a 32x64 C tile as 2x4 WMMA fragments.
// Tiles DMA'd into double-buffered LDS by the Tensor Data Mover (wave 0),
// synchronized with s_wait_tensorcnt + workgroup barrier.
// ---------------------------------------------------------------------------
__global__ __launch_bounds__(256) void gemm_bf16_kernel(
    const unsigned short* __restrict__ A, const unsigned short* __restrict__ W,
    const float* __restrict__ bias, float* __restrict__ outF,
    unsigned short* __restrict__ outB, int M, int N, int K, int doGelu) {
  __shared__ __align__(16) unsigned short As[2][128 * 32];
  __shared__ __align__(16) unsigned short Bs[2][128 * 32];

  const int tid  = threadIdx.x;
  const int lane = tid & 31;
  const int wv   = tid >> 5;
  const int wm   = (wv & 3) * 32;   // wave M offset within tile
  const int wn   = (wv >> 2) * 64;  // wave N offset within tile
  const int bm   = blockIdx.x * 128;
  const int bn   = blockIdx.y * 128;
  const int lr   = lane & 15;
  const int kh   = (lane >> 4) * 8; // element offset of lane's K-chunk

  v8f acc[2][4] = {};

  const int T = K >> 5;                 // number of 32-wide K tiles
  const unsigned row_dw = (unsigned)(K >> 1);

  if (wv == 0) {
    tdm_load_tile(lds_off_u32(&As[0][0]), A + (size_t)bm * K, row_dw, 128u);
    tdm_load_tile(lds_off_u32(&Bs[0][0]), W + (size_t)bn * K, row_dw, 128u);
  }

  for (int t = 0; t < T; ++t) {
    const int buf = t & 1;
    if (wv == 0) {
      if (t + 1 < T) {
        const int k0n = (t + 1) << 5;
        tdm_load_tile(lds_off_u32(&As[buf ^ 1][0]),
                      A + (size_t)bm * K + k0n, row_dw, 128u);
        tdm_load_tile(lds_off_u32(&Bs[buf ^ 1][0]),
                      W + (size_t)bn * K + k0n, row_dw, 128u);
        __builtin_amdgcn_s_wait_tensorcnt(2);  // current tile's 2 DMAs retired
      } else {
        __builtin_amdgcn_s_wait_tensorcnt(0);
      }
    }
    __syncthreads();

    FragBF af[2], bfr[4];
#pragma unroll
    for (int mi = 0; mi < 2; ++mi) {
      const unsigned short* p = &As[buf][(wm + mi * 16 + lr) * 32];
      af[mi].q[0] = *(const u32x4*)(p + kh);
      af[mi].q[1] = *(const u32x4*)(p + kh + 16);
    }
#pragma unroll
    for (int ni = 0; ni < 4; ++ni) {
      const unsigned short* p = &Bs[buf][(wn + ni * 16 + lr) * 32];
      bfr[ni].q[0] = *(const u32x4*)(p + kh);
      bfr[ni].q[1] = *(const u32x4*)(p + kh + 16);
    }
#pragma unroll
    for (int mi = 0; mi < 2; ++mi)
#pragma unroll
      for (int ni = 0; ni < 4; ++ni)
        acc[mi][ni] = __builtin_amdgcn_wmma_f32_16x16x32_bf16(
            false, af[mi].v, false, bfr[ni].v, (short)0, acc[mi][ni], false,
            false);
    __syncthreads();
  }

  // Epilogue: C frag element r -> row = r + 8*(lane>>4), col = lane&15
#pragma unroll
  for (int mi = 0; mi < 2; ++mi) {
#pragma unroll
    for (int ni = 0; ni < 4; ++ni) {
      int col = bn + wn + ni * 16 + lr;
      float bv = bias ? bias[col] : 0.0f;
#pragma unroll
      for (int r = 0; r < 8; ++r) {
        int row = bm + wm + mi * 16 + ((lane >> 4) * 8) + r;
        float vv = acc[mi][ni][r] + bv;
        if (doGelu) vv = gelu_exact(vv);
        size_t idx = (size_t)row * N + col;
        if (outF) outF[idx] = vv;
        if (outB) outB[idx] = f32_to_bf16_bits(vv);
      }
    }
  }
}

// ---------------------------------------------------------------------------
// v [B*S, 1024] (bf16, [b,s,h,d]) -> vT [B,H,DK,S] so attention's B-fragments
// over keys are contiguous.
// ---------------------------------------------------------------------------
__global__ void transpose_v_kernel(const unsigned short* __restrict__ v,
                                   unsigned short* __restrict__ vT) {
  constexpr int S = 2048, H = 16, DK = 64, DM = 1024;
  int i = blockIdx.x * blockDim.x + threadIdx.x;
  if (i < 2 * S * DM) {
    int col = i & (DM - 1);
    int row = i >> 10;            // b*S + s
    int b = row >> 11, s = row & (S - 1);
    int h = col >> 6, d = col & (DK - 1);
    vT[(((size_t)(b * H + h) * DK) + d) * S + s] = v[i];
  }
}

// ---------------------------------------------------------------------------
// Flash attention: block = 128 threads (4 waves), each wave owns 16 q rows.
// grid = (S/64, H, B). Online softmax over 32-key steps; P tile round-trips
// through LDS to convert C-layout -> A-fragment layout.
// ---------------------------------------------------------------------------
__global__ __launch_bounds__(128) void attn_kernel(
    const unsigned short* __restrict__ qb, const unsigned short* __restrict__ kb,
    const unsigned short* __restrict__ vT, const int* __restrict__ mask,
    unsigned short* __restrict__ ctx) {
  constexpr int S = 2048, H = 16, DK = 64, DM = 1024;
  const int lane = threadIdx.x & 31;
  const int wid  = threadIdx.x >> 5;
  const int q0   = blockIdx.x * 64 + wid * 16;
  const int h    = blockIdx.y;
  const int b    = blockIdx.z;
  const int lr   = lane & 15;
  const int kh   = (lane >> 4) * 8;

  __shared__ __align__(16) unsigned short Pt[4][16 * 32];
  unsigned short* P = Pt[wid];

  const unsigned short* qrow = qb + ((size_t)(b * S + q0 + lr)) * DM + h * DK;
  FragBF qf[2];
#pragma unroll
  for (int c = 0; c < 2; ++c) {
    qf[c].q[0] = *(const u32x4*)(qrow + c * 32 + kh);
    qf[c].q[1] = *(const u32x4*)(qrow + c * 32 + kh + 16);
  }

  float rmax[8], rsum[8];
  v8f acc[4] = {};
#pragma unroll
  for (int r = 0; r < 8; ++r) { rmax[r] = -1e30f; rsum[r] = 0.0f; }
  const float scale = 0.125f;  // 1/sqrt(64)

  for (int kk = 0; kk < S; kk += 32) {
    v8f s0 = {}, s1 = {};
    const unsigned short* krow0 =
        kb + ((size_t)(b * S + kk + lr)) * DM + h * DK;
    const unsigned short* krow1 = krow0 + (size_t)16 * DM;
#pragma unroll
    for (int c = 0; c < 2; ++c) {
      FragBF kf0, kf1;
      kf0.q[0] = *(const u32x4*)(krow0 + c * 32 + kh);
      kf0.q[1] = *(const u32x4*)(krow0 + c * 32 + kh + 16);
      kf1.q[0] = *(const u32x4*)(krow1 + c * 32 + kh);
      kf1.q[1] = *(const u32x4*)(krow1 + c * 32 + kh + 16);
      s0 = __builtin_amdgcn_wmma_f32_16x16x32_bf16(false, qf[c].v, false,
                                                   kf0.v, (short)0, s0, false,
                                                   false);
      s1 = __builtin_amdgcn_wmma_f32_16x16x32_bf16(false, qf[c].v, false,
                                                   kf1.v, (short)0, s1, false,
                                                   false);
    }
    const int m0 = mask[b * S + kk + lr];
    const int m1 = mask[b * S + kk + 16 + lr];
    float sc0[8], sc1[8], tmax[8];
#pragma unroll
    for (int r = 0; r < 8; ++r) {
      sc0[r] = m0 ? s0[r] * scale : -1e30f;
      sc1[r] = m1 ? s1[r] * scale : -1e30f;
      tmax[r] = fmaxf(sc0[r], sc1[r]);
    }
#pragma unroll
    for (int off = 8; off >= 1; off >>= 1)
#pragma unroll
      for (int r = 0; r < 8; ++r)
        tmax[r] = fmaxf(tmax[r], __shfl_xor(tmax[r], off));

    float alpha[8], p0[8], p1[8], tsum[8];
#pragma unroll
    for (int r = 0; r < 8; ++r) {
      float mnew = fmaxf(rmax[r], tmax[r]);
      alpha[r] = __expf(rmax[r] - mnew);
      rmax[r] = mnew;
      p0[r] = __expf(sc0[r] - mnew);
      p1[r] = __expf(sc1[r] - mnew);
      tsum[r] = p0[r] + p1[r];
    }
#pragma unroll
    for (int off = 8; off >= 1; off >>= 1)
#pragma unroll
      for (int r = 0; r < 8; ++r) tsum[r] += __shfl_xor(tsum[r], off);
#pragma unroll
    for (int r = 0; r < 8; ++r) rsum[r] = rsum[r] * alpha[r] + tsum[r];
#pragma unroll
    for (int d = 0; d < 4; ++d)
#pragma unroll
      for (int r = 0; r < 8; ++r) acc[d][r] *= alpha[r];

#pragma unroll
    for (int r = 0; r < 8; ++r) {
      int m = r + ((lane >> 4) * 8);
      P[m * 32 + lr] = f32_to_bf16_bits(p0[r]);
      P[m * 32 + lr + 16] = f32_to_bf16_bits(p1[r]);
    }
    __syncthreads();
    FragBF pf;
    pf.q[0] = *(const u32x4*)(P + lr * 32 + kh);
    pf.q[1] = *(const u32x4*)(P + lr * 32 + kh + 16);
#pragma unroll
    for (int d = 0; d < 4; ++d) {
      const unsigned short* vrow =
          vT + (((size_t)(b * H + h) * DK) + d * 16 + lr) * S + kk;
      FragBF vf;
      vf.q[0] = *(const u32x4*)(vrow + kh);
      vf.q[1] = *(const u32x4*)(vrow + kh + 16);
      acc[d] = __builtin_amdgcn_wmma_f32_16x16x32_bf16(
          false, pf.v, false, vf.v, (short)0, acc[d], false, false);
    }
    __syncthreads();
  }

#pragma unroll
  for (int d = 0; d < 4; ++d)
#pragma unroll
    for (int r = 0; r < 8; ++r) {
      int m = r + ((lane >> 4) * 8);
      float o = acc[d][r] / rsum[r];
      ctx[((size_t)(b * S + q0 + m)) * DM + h * DK + d * 16 + lr] =
          f32_to_bf16_bits(o);
    }
}

// ---------------------------------------------------------------------------
// out = LayerNorm(a + b) * g + beta ; optional bf16 copy. One block per row.
// ---------------------------------------------------------------------------
__global__ __launch_bounds__(256) void add_ln_kernel(
    const float* __restrict__ a, const float* __restrict__ b,
    const float* __restrict__ g, const float* __restrict__ beta,
    float* __restrict__ outF, unsigned short* __restrict__ outB) {
  constexpr int D = 1024;
  const int row = blockIdx.x;
  const int tid = threadIdx.x;
  const float* ar = a + (size_t)row * D;
  const float* br = b + (size_t)row * D;
  __shared__ float red[16];
  float vals[4];
  float s = 0.f, s2 = 0.f;
#pragma unroll
  for (int j = 0; j < 4; ++j) {
    float v = ar[tid + j * 256] + br[tid + j * 256];
    vals[j] = v;
    s += v;
    s2 += v * v;
  }
#pragma unroll
  for (int off = 16; off >= 1; off >>= 1) {
    s += __shfl_xor(s, off);
    s2 += __shfl_xor(s2, off);
  }
  if ((tid & 31) == 0) {
    red[tid >> 5] = s;
    red[8 + (tid >> 5)] = s2;
  }
  __syncthreads();
  float ts = 0.f, ts2 = 0.f;
#pragma unroll
  for (int w = 0; w < 8; ++w) {
    ts += red[w];
    ts2 += red[8 + w];
  }
  const float mu = ts * (1.0f / D);
  const float var = ts2 * (1.0f / D) - mu * mu;
  const float inv = rsqrtf(var + 1e-5f);
#pragma unroll
  for (int j = 0; j < 4; ++j) {
    int c = tid + j * 256;
    float o = (vals[j] - mu) * inv * g[c] + beta[c];
    outF[(size_t)row * D + c] = o;
    if (outB) outB[(size_t)row * D + c] = f32_to_bf16_bits(o);
  }
}

// ---------------------------------------------------------------------------
extern "C" void kernel_launch(void* const* d_in, const int* in_sizes, int n_in,
                              void* d_out, int out_size, void* d_ws,
                              size_t ws_size, hipStream_t stream) {
  (void)in_sizes; (void)n_in; (void)out_size; (void)ws_size;
  constexpr int D = 1024, H = 16, FH = 4096, B = 2, S = 2048;
  constexpr int M = B * S;  // 4096 rows

  const float* x    = (const float*)d_in[0];
  const int*   mask = (const int*)d_in[1];
  const float* wq = (const float*)d_in[2];
  const float* bq = (const float*)d_in[3];
  const float* wk = (const float*)d_in[4];
  const float* bk = (const float*)d_in[5];
  const float* wv = (const float*)d_in[6];
  const float* bv = (const float*)d_in[7];
  const float* wo = (const float*)d_in[8];
  const float* bo = (const float*)d_in[9];
  const float* ln_g = (const float*)d_in[10];
  const float* ln_b = (const float*)d_in[11];
  const float* w1 = (const float*)d_in[12];
  const float* b1 = (const float*)d_in[13];
  const float* w2 = (const float*)d_in[14];
  const float* b2 = (const float*)d_in[15];
  float* out = (float*)d_out;

  char* base = (char*)d_ws;
  size_t off = 0;
  auto carve = [&](size_t bytes) -> char* {
    char* p = base + off;
    off += (bytes + 255) & ~(size_t)255;
    return p;
  };
  unsigned short* wqb = (unsigned short*)carve((size_t)D * D * 2);
  unsigned short* wkb = (unsigned short*)carve((size_t)D * D * 2);
  unsigned short* wvb = (unsigned short*)carve((size_t)D * D * 2);
  unsigned short* wob = (unsigned short*)carve((size_t)D * D * 2);
  unsigned short* w1b = (unsigned short*)carve((size_t)FH * D * 2);
  unsigned short* w2b = (unsigned short*)carve((size_t)D * FH * 2);
  unsigned short* xb  = (unsigned short*)carve((size_t)M * D * 2);
  unsigned short* qb  = (unsigned short*)carve((size_t)M * D * 2);
  unsigned short* kb  = (unsigned short*)carve((size_t)M * D * 2);
  unsigned short* vb  = (unsigned short*)carve((size_t)M * D * 2);
  unsigned short* vTb = (unsigned short*)carve((size_t)M * D * 2);
  float* attnF        = (float*)carve((size_t)M * D * 4);
  float* out1F        = (float*)carve((size_t)M * D * 4);
  unsigned short* o1b = (unsigned short*)carve((size_t)M * D * 2);
  unsigned short* hB  = (unsigned short*)carve((size_t)M * FH * 2);
  unsigned short* ctxb = vb;   // v consumed by transpose before attention
  float* ffnF = attnF;         // attn_out consumed by first add_ln

  auto cdiv = [](int a, int b) { return (a + b - 1) / b; };

  // 1) fp32 -> bf16 conversions
  cvt_kernel<<<cdiv(M * D, 256), 256, 0, stream>>>(x, xb, M * D);
  cvt_kernel<<<cdiv(D * D, 256), 256, 0, stream>>>(wq, wqb, D * D);
  cvt_kernel<<<cdiv(D * D, 256), 256, 0, stream>>>(wk, wkb, D * D);
  cvt_kernel<<<cdiv(D * D, 256), 256, 0, stream>>>(wv, wvb, D * D);
  cvt_kernel<<<cdiv(D * D, 256), 256, 0, stream>>>(wo, wob, D * D);
  cvt_kernel<<<cdiv(FH * D, 256), 256, 0, stream>>>(w1, w1b, FH * D);
  cvt_kernel<<<cdiv(D * FH, 256), 256, 0, stream>>>(w2, w2b, D * FH);

  // 2) QKV projections (TDM + bf16 WMMA GEMMs)
  dim3 gD(M / 128, D / 128);
  gemm_bf16_kernel<<<gD, 256, 0, stream>>>(xb, wqb, bq, nullptr, qb, M, D, D, 0);
  gemm_bf16_kernel<<<gD, 256, 0, stream>>>(xb, wkb, bk, nullptr, kb, M, D, D, 0);
  gemm_bf16_kernel<<<gD, 256, 0, stream>>>(xb, wvb, bv, nullptr, vb, M, D, D, 0);

  // 3) v -> vT for contiguous attention B-fragments
  transpose_v_kernel<<<cdiv(M * D, 256), 256, 0, stream>>>(vb, vTb);

  // 4) Flash attention
  attn_kernel<<<dim3(S / 64, H, B), 128, 0, stream>>>(qb, kb, vTb, mask, ctxb);

  // 5) Output projection -> fp32
  gemm_bf16_kernel<<<gD, 256, 0, stream>>>(ctxb, wob, bo, attnF, nullptr, M, D,
                                           D, 0);

  // 6) out1 = LN(x + attn_out); also bf16 for FFN
  add_ln_kernel<<<M, 256, 0, stream>>>(x, attnF, ln_g, ln_b, out1F, o1b);

  // 7) FFN up + exact GELU (bf16 out)
  dim3 gF(M / 128, FH / 128);
  gemm_bf16_kernel<<<gF, 256, 0, stream>>>(o1b, w1b, b1, nullptr, hB, M, FH, D,
                                           1);

  // 8) FFN down -> fp32 (reuses attnF)
  gemm_bf16_kernel<<<gD, 256, 0, stream>>>(hB, w2b, b2, ffnF, nullptr, M, D,
                                           FH, 0);

  // 9) out2 = LN(out1 + ffn) -> d_out
  add_ln_kernel<<<M, 256, 0, stream>>>(out1F, ffnF, ln_g, ln_b, out, nullptr);
}
